// WAEEncoder_8701603741791
// MI455X (gfx1250) — matrix operand order repaired and checked by reading, not compiled, not run on hardware
//
#include <hip/hip_runtime.h>
#include <hip/hip_bf16.h>

typedef __attribute__((ext_vector_type(16))) _Float16 v16h;
typedef __attribute__((ext_vector_type(8)))  float    v8f;

#define LTOK 50
#define HDIM 100
#define ODIM 50
#define NT1  7              // N tiles for GEMM1: 7*16 = 112 >= 100
#define NT2  4              // N tiles for GEMM2: 4*16 = 64  >= 50
#define NWAVE 8
#define ROWS_PER_WAVE 16
#define ROWS_PER_BLOCK (NWAVE * ROWS_PER_WAVE)

__device__ __forceinline__ float softplus_f(float x) {
    // numerically stable log1p(exp(x))
    return fmaxf(x, 0.0f) + log1pf(expf(-fabsf(x)));
}

// Map a K-position c (0..31 within a 32-wide chunk) of row m to the f16 A-matrix
// fragment slot (lane', element) per CDNA5 ISA 16-bit A 16x32 layout.
__device__ __forceinline__ void afrag_slot(int m, int c, int& lanep, int& e) {
    int hi = (c >> 3) & 1;                 // K 8..15 / 24..31 live in lanes 16..31
    e      = (c & 7) + ((c >> 4) & 1) * 8; // K 16..23 / 24..31 are elements 8..15
    lanep  = m + hi * 16;
}

__global__ __launch_bounds__(256) void wae_fused_kernel(
    const int*   __restrict__ toks_g,
    const float* __restrict__ W0,
    const float* __restrict__ b0,
    const float* __restrict__ W1,
    const float* __restrict__ b1,
    const float* __restrict__ W2,
    const float* __restrict__ b2,
    float*       __restrict__ out,
    int Btotal)
{
    // B-fragment-swizzled weight tiles (shared by all 8 waves)
    __shared__ __align__(32) _Float16 w1f[NT1 * 4 * 512];   // 28 frags * 512 halves
    __shared__ __align__(32) _Float16 w2f[NT2 * 4 * 512];   // 16 frags * 512 halves
    // per-wave A-fragment buffers (4 K-chunks x 512 halves), pre-zeroed for padding
    __shared__ __align__(32) _Float16 a0buf[NWAVE][4 * 512];  // h0 (also reused as f32 logits)
    __shared__ __align__(32) _Float16 a1buf[NWAVE][4 * 512];  // h1
    __shared__ int   toks[NWAVE][64];
    __shared__ float rowmax[NWAVE][ROWS_PER_WAVE];
    __shared__ float rowsum[NWAVE][ROWS_PER_WAVE];

    const int tid  = threadIdx.x;
    const int lane = tid & 31;
    const int wv   = tid >> 5;

    // ---- Stage W1 into B-fragment layout: frag f=(nt*4+kk), lane ln, elem e ----
    // B 32x16 f16 layout: lane = N (lanes 16..31 carry K 8..15 / 24..31), elems 8..15 = K+16
    for (int idx = tid; idx < NT1 * 4 * 512; idx += 256) {
        int f  = idx >> 9, r = idx & 511;
        int ln = r >> 4,  e = r & 15;
        int nt = f >> 2,  kk = f & 3;
        int col = nt * 16 + (ln & 15);
        int k   = kk * 32 + ((ln >> 4) << 3) + (e & 7) + ((e >> 3) << 4);
        float v = (k < HDIM && col < HDIM) ? W1[k * HDIM + col] : 0.0f;
        w1f[idx] = (_Float16)v;
    }
    for (int idx = tid; idx < NT2 * 4 * 512; idx += 256) {
        int f  = idx >> 9, r = idx & 511;
        int ln = r >> 4,  e = r & 15;
        int nt = f >> 2,  kk = f & 3;
        int col = nt * 16 + (ln & 15);
        int k   = kk * 32 + ((ln >> 4) << 3) + (e & 7) + ((e >> 3) << 4);
        float v = (k < HDIM && col < ODIM) ? W2[k * ODIM + col] : 0.0f;
        w2f[idx] = (_Float16)v;
    }
    // zero the A buffers (provides K=100..127 zero padding)
    {
        uint32_t* z0 = (uint32_t*)a0buf[wv];
        uint32_t* z1 = (uint32_t*)a1buf[wv];
        for (int i = lane; i < 1024; i += 32) { z0[i] = 0u; z1[i] = 0u; }
    }
    __syncthreads();

    const int rowbase = blockIdx.x * ROWS_PER_BLOCK + wv * ROWS_PER_WAVE;

    // ---- Phase 1: multi-hot gather (with dedupe) + bias + softplus -> a0 frags ----
    for (int r = 0; r < ROWS_PER_WAVE; ++r) {
        const int rowg  = rowbase + r;
        const bool valid = rowg < Btotal;
        const int* src = toks_g + (long long)rowg * LTOK;
        const int t0 = lane, t1 = lane + 32;
        toks[wv][t0] = (valid && t0 < LTOK) ? src[t0] : -1;
        toks[wv][t1] = (valid && t1 < LTOK) ? src[t1] : -1;
        __syncthreads();

        // dedupe: duplicates must contribute only once (multi-hot "set" semantics).
        // First occurrences are never overwritten, so concurrent -1 marking is safe.
        int my0 = toks[wv][t0], my1 = toks[wv][t1];
        bool d0 = false, d1 = false;
        if (my0 >= 0) for (int tp = 0; tp < t0; ++tp) if (toks[wv][tp] == my0) { d0 = true; break; }
        if (my1 >= 0) for (int tp = 0; tp < t1; ++tp) if (toks[wv][tp] == my1) { d1 = true; break; }
        if (d0) toks[wv][t0] = -1;
        if (d1) toks[wv][t1] = -1;
        __syncthreads();

        // 25 lanes * float4 = 100 columns of W0 gathered per surviving token
        float4 acc = make_float4(0.f, 0.f, 0.f, 0.f);
        for (int t = 0; t < LTOK; ++t) {
            int tok = toks[wv][t];                 // wave-uniform broadcast
            if (tok >= 0 && lane < 25) {
                const float4 w = *(const float4*)(W0 + (long long)tok * HDIM + lane * 4);
                acc.x += w.x; acc.y += w.y; acc.z += w.z; acc.w += w.w;
            }
        }
        if (lane < 25) {
            float vals[4] = {acc.x, acc.y, acc.z, acc.w};
#pragma unroll
            for (int q = 0; q < 4; ++q) {
                int c = lane * 4 + q;              // column 0..99
                float h = softplus_f(vals[q] + b0[c]);
                int kk = c >> 5, cc = c & 31, lp, e;
                afrag_slot(r, cc, lp, e);
                a0buf[wv][kk * 512 + lp * 16 + e] = (_Float16)h;
            }
        }
        __syncthreads();
    }

    // ---- GEMM1: h0[16x128] @ W1[128x112] via v_wmma_f32_16x16x32_f16 ----
    v16h A0 = *(const v16h*)&a0buf[wv][0 * 512 + lane * 16];
    v16h A1 = *(const v16h*)&a0buf[wv][1 * 512 + lane * 16];
    v16h A2 = *(const v16h*)&a0buf[wv][2 * 512 + lane * 16];
    v16h A3 = *(const v16h*)&a0buf[wv][3 * 512 + lane * 16];

#pragma unroll
    for (int nt = 0; nt < NT1; ++nt) {
        v8f acc = {};
        {
            v16h Bf;
            Bf  = *(const v16h*)&w1f[(nt * 4 + 0) * 512 + lane * 16];
            acc = __builtin_amdgcn_wmma_f32_16x16x32_f16(false, A0, false, Bf, (short)0, acc, false, false);
            Bf  = *(const v16h*)&w1f[(nt * 4 + 1) * 512 + lane * 16];
            acc = __builtin_amdgcn_wmma_f32_16x16x32_f16(false, A1, false, Bf, (short)0, acc, false, false);
            Bf  = *(const v16h*)&w1f[(nt * 4 + 2) * 512 + lane * 16];
            acc = __builtin_amdgcn_wmma_f32_16x16x32_f16(false, A2, false, Bf, (short)0, acc, false, false);
            Bf  = *(const v16h*)&w1f[(nt * 4 + 3) * 512 + lane * 16];
            acc = __builtin_amdgcn_wmma_f32_16x16x32_f16(false, A3, false, Bf, (short)0, acc, false, false);
        }
        // C/D f32 layout: lane%16 = N, VGPR j = M row (lanes>=16 carry M+8)
        int n = nt * 16 + (lane & 15);
        float bias = (n < HDIM) ? b1[n] : 0.0f;
#pragma unroll
        for (int j = 0; j < 8; ++j) {
            int m = j + ((lane >> 4) << 3);
            float v = softplus_f(acc[j] + bias);
            if (n < HDIM) {
                int kk = n >> 5, cc = n & 31, lp, e;
                afrag_slot(m, cc, lp, e);
                a1buf[wv][kk * 512 + lp * 16 + e] = (_Float16)v;
            }
        }
    }
    __syncthreads();

    // ---- GEMM2: h1[16x128] @ W2[128x64] -> logits (f32, into a0buf alias) ----
    v16h C0 = *(const v16h*)&a1buf[wv][0 * 512 + lane * 16];
    v16h C1 = *(const v16h*)&a1buf[wv][1 * 512 + lane * 16];
    v16h C2 = *(const v16h*)&a1buf[wv][2 * 512 + lane * 16];
    v16h C3 = *(const v16h*)&a1buf[wv][3 * 512 + lane * 16];

    float* logits = (float*)a0buf[wv];   // 16 x 64 f32 tile (4096 B, exact alias)

#pragma unroll
    for (int nt = 0; nt < NT2; ++nt) {
        v8f acc = {};
        {
            v16h Bf;
            Bf  = *(const v16h*)&w2f[(nt * 4 + 0) * 512 + lane * 16];
            acc = __builtin_amdgcn_wmma_f32_16x16x32_f16(false, C0, false, Bf, (short)0, acc, false, false);
            Bf  = *(const v16h*)&w2f[(nt * 4 + 1) * 512 + lane * 16];
            acc = __builtin_amdgcn_wmma_f32_16x16x32_f16(false, C1, false, Bf, (short)0, acc, false, false);
            Bf  = *(const v16h*)&w2f[(nt * 4 + 2) * 512 + lane * 16];
            acc = __builtin_amdgcn_wmma_f32_16x16x32_f16(false, C2, false, Bf, (short)0, acc, false, false);
            Bf  = *(const v16h*)&w2f[(nt * 4 + 3) * 512 + lane * 16];
            acc = __builtin_amdgcn_wmma_f32_16x16x32_f16(false, C3, false, Bf, (short)0, acc, false, false);
        }
        int n = nt * 16 + (lane & 15);
        float bias = (n < ODIM) ? b2[n] : 0.0f;
#pragma unroll
        for (int j = 0; j < 8; ++j) {
            int m = j + ((lane >> 4) << 3);
            logits[m * 64 + n] = acc[j] + bias;   // n>=50 region is never read
        }
    }
    __syncthreads();

    // ---- Softmax over 50 logits per row ----
    if (lane < ROWS_PER_WAVE) {
        float mx = -INFINITY;
        for (int c = 0; c < ODIM; ++c) mx = fmaxf(mx, logits[lane * 64 + c]);
        float s = 0.0f;
        for (int c = 0; c < ODIM; ++c) s += expf(logits[lane * 64 + c] - mx);
        rowmax[wv][lane] = mx;
        rowsum[wv][lane] = s;
    }
    __syncthreads();

    for (int idx = lane; idx < ROWS_PER_WAVE * ODIM; idx += 32) {
        int rr = idx / ODIM, c = idx - rr * ODIM;
        int rowg = rowbase + rr;
        if (rowg < Btotal) {
            float v = expf(logits[rr * 64 + c] - rowmax[wv][rr]) / rowsum[wv][rr];
            out[(long long)rowg * ODIM + c] = v;
        }
    }
}

extern "C" void kernel_launch(void* const* d_in, const int* in_sizes, int n_in,
                              void* d_out, int out_size, void* d_ws, size_t ws_size,
                              hipStream_t stream) {
    const int*   toks = (const int*)  d_in[0];
    const float* W0   = (const float*)d_in[1];
    const float* b0   = (const float*)d_in[2];
    const float* W1   = (const float*)d_in[3];
    const float* b1   = (const float*)d_in[4];
    const float* W2   = (const float*)d_in[5];
    const float* b2   = (const float*)d_in[6];
    float* out = (float*)d_out;

    const int Btotal = in_sizes[0] / LTOK;   // 32768
    const int nblocks = (Btotal + ROWS_PER_BLOCK - 1) / ROWS_PER_BLOCK;

    wae_fused_kernel<<<nblocks, 256, 0, stream>>>(toks, W0, b0, W1, b1, W2, b2, out, Btotal);
}
